// AgnosticResidualInteractionBlock_65017214927001
// MI455X (gfx1250) — matrix operand order repaired and checked by reading, not compiled, not run on hardware
//
#include <hip/hip_runtime.h>
#include <hip/hip_bf16.h>
#include <math.h>

// ---------------------------------------------------------------------------
// MACE-style interaction block for MI455X (gfx1250), wave32 + WMMA bf16.
// All dense contractions -> v_wmma_f32_16x16x32_bf16, operands staged as
// bf16 (A row-major, B transposed/col-major, K padded to mult. of 32) so
// every fragment is built from contiguous b128 loads (global or LDS).
// Edge scatter -> global_atomic_add_f32. N, E multiples of 16 => no guards.
// ---------------------------------------------------------------------------

typedef __attribute__((ext_vector_type(16))) __bf16 v16bf;
typedef __attribute__((ext_vector_type(8)))  __bf16 bf8v;
typedef __attribute__((ext_vector_type(8)))  float  v8f;

__device__ __forceinline__ __bf16 f2bf(float x) {
    union { float f; unsigned u; } v; v.f = x;
    unsigned r = v.u + 0x7FFFu + ((v.u >> 16) & 1u);   // RNE
    return __builtin_bit_cast(__bf16, (unsigned short)(r >> 16));
}
__device__ __forceinline__ float bf2f(__bf16 x) {
    union { unsigned u; float f; } v;
    v.u = ((unsigned)__builtin_bit_cast(unsigned short, x)) << 16;
    return v.f;
}

// A-fragment (16x32, 16-bit, ISA 7.12.2): lane holds row lane&15;
//   elements 0..7  = K (k0 + koff + 0..7),  koff = lane<16 ? 0 : 8
//   elements 8..15 = K (k0 + 16 + koff + 0..7)
// B-fragment (32x16): lane holds col lane&15; same K order as A:
//   elements 0..7 = K (k0 + kbb + 0..7), 8..15 = K (k0 + 16 + kbb + 0..7), kbb = lane<16 ? 0 : 8
__device__ __forceinline__ v16bf cat8(bf8v lo, bf8v hi) {
    return __builtin_shufflevector(lo, hi, 0,1,2,3,4,5,6,7,8,9,10,11,12,13,14,15);
}
__device__ __forceinline__ v8f wmma_bf16(v16bf a, v16bf b, v8f c) {
    return __builtin_amdgcn_wmma_f32_16x16x32_bf16(false, a, false, b,
                                                   (short)0, c, false, false);
}

// ---------------------------------------------------------------------------
// Prep kernels (once per launch; ~50MB total traffic, negligible at 23.3TB/s)
// ---------------------------------------------------------------------------
__global__ void f32_to_bf16_kernel(const float* __restrict__ src,
                                   __bf16* __restrict__ dst, long n) {
    long i = (long)blockIdx.x * blockDim.x + threadIdx.x;
    if (i < n) dst[i] = f2bf(src[i]);
}
__global__ void pad_rows_bf16_kernel(const float* __restrict__ src,
                                     __bf16* __restrict__ dst,
                                     long rows, int incols, int outcols) {
    long i = (long)blockIdx.x * blockDim.x + threadIdx.x;
    if (i >= rows * (long)outcols) return;
    long r = i / outcols; int k = (int)(i - r * outcols);
    dst[i] = (k < incols) ? f2bf(src[r * incols + k]) : f2bf(0.0f);
}
// (N,C,3) f32 -> 3 planes of (N,C) bf16
__global__ void planarize_nfv_kernel(const float* __restrict__ src,
                                     __bf16* __restrict__ dst, int Nn) {
    long i = (long)blockIdx.x * blockDim.x + threadIdx.x;
    long plane_sz = (long)Nn * 128;
    if (i >= 3 * plane_sz) return;
    int plane = (int)(i / plane_sz);
    long rem = i - plane * plane_sz;
    int n = (int)(rem >> 7), c = (int)(rem & 127);
    dst[i] = f2bf(src[(long)n * 384 + c * 3 + plane]);
}
// W (K x Nout) f32 -> Wt (Nout x Kpad) bf16, zero-padded K
__global__ void transpose_w_kernel(const float* __restrict__ src,
                                   __bf16* __restrict__ dst,
                                   int K, int Nout, int Kpad) {
    long i = (long)blockIdx.x * blockDim.x + threadIdx.x;
    if (i >= (long)Nout * Kpad) return;
    int col = (int)(i / Kpad), k = (int)(i - (long)col * Kpad);
    dst[i] = (k < K) ? f2bf(src[(long)k * Nout + col]) : f2bf(0.0f);
}

// ---------------------------------------------------------------------------
// GEMM: C = op(scale * A @ B), A bf16 (M x K row-major), Bt bf16 (Nout x Kpad
// col-major). One wave = one 16x16 tile. Inner loop: 4x global b128 + 1 wmma.
// ---------------------------------------------------------------------------
template <int KSTEPS, bool OUT_BF16, bool SILU>
__global__ void gemm_bf16t_kernel(const __bf16* __restrict__ A, int lda,
                                  const __bf16* __restrict__ Bt, int ldbt,
                                  float* __restrict__ Cf, __bf16* __restrict__ Cb,
                                  int ldc, int cstride, float scale) {
    const int lane = threadIdx.x;
    const int am   = lane & 15;
    const int row  = blockIdx.x * 16 + am;
    const int col  = blockIdx.y * 16 + am;
    const int koff = (lane < 16) ? 0 : 8;
    const int kbb  = (lane < 16) ? 0 : 8;
    const __bf16* pa = A  + (long)row * lda  + koff;
    const __bf16* pb = Bt + (long)col * ldbt + kbb;

    v8f acc = {};
#pragma unroll
    for (int ks = 0; ks < KSTEPS; ++ks) {
        const int k0 = ks * 32;
        v16bf a = cat8(*(const bf8v*)(pa + k0), *(const bf8v*)(pa + k0 + 16));
        v16bf b = cat8(*(const bf8v*)(pb + k0), *(const bf8v*)(pb + k0 + 16));
        acc = wmma_bf16(a, b, acc);
    }

    const int rb = (lane < 16) ? 0 : 8;     // D: VGPR r -> row rb+r, col lane&15
#pragma unroll
    for (int r = 0; r < 8; ++r) {
        long orow = blockIdx.x * 16 + rb + r;
        float v = acc[r] * scale;
        if (SILU) v = v / (1.0f + __expf(-v));
        if (OUT_BF16) Cb[orow * ldc + col] = f2bf(v);
        else          Cf[orow * ldc + (long)col * cstride] = v;
    }
}

// ---------------------------------------------------------------------------
// Self-connection tensor product, K = C*A = 1280:
//   z==0: sc_s[n,o]   = inv * sum_{k=c*10+a} s[n,c]*attr[n,a]*W_sc_s[k,o]
//   z>=1: sc_v[n,o,i] = inv * sum_k v[n,c,i]*attr[n,a]*W_sc_v[k,o], i=z-1
// Block = 256 threads = 8 waves. The 16x1280 tp tile is built ONCE in LDS
// (bf16, fragment-ready), shared by all 8 column-tile waves; inner loop is
// then 2x ds_load_b128 (A) + 2x global b128 (B) + 1 wmma per k-step.
// ---------------------------------------------------------------------------
__global__ void sc_tp_kernel(const float* __restrict__ attrs,     // N x 10
                             const float* __restrict__ feats_s,   // N x 128
                             const float* __restrict__ feats_v,   // N x 128 x 3
                             const __bf16* __restrict__ Wt_sc_s,  // 128 x 1280
                             const __bf16* __restrict__ Wt_sc_v,  // 128 x 1280
                             float* __restrict__ sc_out,          // N x 512
                             int Nn) {
    __shared__ __bf16 s_tp[16 * 1280];      // 40KB, row-major (m, k)
    __shared__ float  s_feat[16][128];
    __shared__ float  s_attr[16][10];

    const int tid   = threadIdx.x;
    const int z     = blockIdx.z;           // 0 = scalar, 1..3 = vector comp
    const int comp  = z - 1;
    const int nbase = blockIdx.x * 16;

    // stage node rows
    for (int t = tid; t < 16 * 128; t += 256) {
        int m = t >> 7, c = t & 127;
        int node = nbase + m;
        s_feat[m][c] = (z == 0) ? feats_s[(long)node * 128 + c]
                                : feats_v[(long)node * 384 + c * 3 + comp];
    }
    for (int t = tid; t < 160; t += 256) {
        int m = t / 10, a = t - m * 10;
        s_attr[m][a] = attrs[(long)(nbase + m) * 10 + a];
    }
    __syncthreads();

    // build tp tile once: tp[m][k] = feat[m][k/10] * attr[m][k%10]
    for (unsigned idx = tid; idx < 16u * 1280u; idx += 256u) {
        unsigned m = idx / 1280u;
        unsigned k = idx - m * 1280u;
        unsigned c = k / 10u;
        unsigned a = k - c * 10u;
        s_tp[idx] = f2bf(s_feat[m][c] * s_attr[m][a]);
    }
    __syncthreads();

    // 8 waves x one 16-col tile each
    const int lane = tid & 31;
    const int wav  = tid >> 5;
    const int am   = lane & 15;
    const int koff = (lane < 16) ? 0 : 8;
    const int kbb  = (lane < 16) ? 0 : 8;
    const int colB = wav * 16 + am;
    const __bf16* Wt = (z == 0) ? Wt_sc_s : Wt_sc_v;
    const __bf16* pb = Wt + (long)colB * 1280 + kbb;
    const __bf16* pa = s_tp + am * 1280 + koff;

    v8f acc = {};
#pragma unroll 4
    for (int ks = 0; ks < 40; ++ks) {
        const int k0 = ks * 32;
        v16bf a = cat8(*(const bf8v*)(pa + k0), *(const bf8v*)(pa + k0 + 16));
        v16bf b = cat8(*(const bf8v*)(pb + k0), *(const bf8v*)(pb + k0 + 16));
        acc = wmma_bf16(a, b, acc);
    }

    const float inv = 0.02795084972f;       // 1/sqrt(1280)
    const int rb = (lane < 16) ? 0 : 8;
#pragma unroll
    for (int r = 0; r < 8; ++r) {
        long node = nbase + rb + r;
        float v = acc[r] * inv;
        long off = (z == 0) ? (node * 512 + colB)
                            : (node * 512 + 128 + colB * 3 + comp);
        sc_out[off] = v;
    }
}

// ---------------------------------------------------------------------------
// Fused edge kernel: h3 (bf16, E x 64) -> tpw 16x16 tiles (WMMA, K=64, never
// stored) -> message terms -> atomic scatter into msg_s / planar msg_v.
// One wave handles 16 edges.
// ---------------------------------------------------------------------------
__global__ void edge_msg_kernel(const __bf16* __restrict__ h3,       // E x 64
                                const __bf16* __restrict__ Wt_w3,    // 640 x 64
                                const float*  __restrict__ edge_attrs, // E x 4
                                const int*    __restrict__ senders,
                                const int*    __restrict__ receivers,
                                const __bf16* __restrict__ xs_bf,    // N x 128
                                const __bf16* __restrict__ xv_bf,    // 3 x N x 128
                                float* __restrict__ msg_s,           // N x 256
                                float* __restrict__ msg_v,           // 3 x N x 384
                                int Nn) {
    __shared__ float s_xs[16][128];
    __shared__ float s_xv[3][16][128];
    __shared__ float s_es[16];
    __shared__ float s_ev[16][3];
    __shared__ int   s_send[16];
    __shared__ int   s_recv[16];

    const int lane  = threadIdx.x;
    const int ebase = blockIdx.x * 16;
    const long plane_sz = (long)Nn * 128;
    const long vplane   = (long)Nn * 384;

    if (lane < 16) {
        int e = ebase + lane;
        s_es[lane]    = edge_attrs[(long)e * 4 + 0];
        s_ev[lane][0] = edge_attrs[(long)e * 4 + 1];
        s_ev[lane][1] = edge_attrs[(long)e * 4 + 2];
        s_ev[lane][2] = edge_attrs[(long)e * 4 + 3];
        s_send[lane]  = senders[e];
        s_recv[lane]  = receivers[e];
    }
    __syncthreads();

    for (int t = lane; t < 16 * 128; t += 32) {
        int m = t >> 7, c = t & 127;
        long sb = (long)s_send[m] * 128 + c;
        s_xs[m][c]    = bf2f(xs_bf[sb]);
        s_xv[0][m][c] = bf2f(xv_bf[sb]);
        s_xv[1][m][c] = bf2f(xv_bf[plane_sz + sb]);
        s_xv[2][m][c] = bf2f(xv_bf[2 * plane_sz + sb]);
    }
    __syncthreads();

    // A tiles of h3 (16 edges x 64): two 16x32 k-slices, built once.
    const int am   = lane & 15;
    const int koff = (lane < 16) ? 0 : 8;
    const int kbb  = (lane < 16) ? 0 : 8;
    const __bf16* pa = h3 + (long)(ebase + am) * 64 + koff;
    v16bf a0 = cat8(*(const bf8v*)(pa),      *(const bf8v*)(pa + 16));
    v16bf a1 = cat8(*(const bf8v*)(pa + 32), *(const bf8v*)(pa + 48));

    const float invH = 0.125f;               // 1/sqrt(64)
    const float is3  = 0.57735026919f;       // 1/sqrt(3)
    const float is2  = 0.70710678119f;       // 1/sqrt(2)
    const int   rb   = (lane < 16) ? 0 : 8;

#pragma unroll
    for (int j = 0; j < 5; ++j) {            // w1..w5 chunks of tpw
#pragma unroll
        for (int t = 0; t < 8; ++t) {        // 16-col tiles within a chunk
            const int c0   = t * 16;
            const int colW = j * 128 + c0 + am;
            const __bf16* pb = Wt_w3 + (long)colW * 64 + kbb;
            v16bf b0 = cat8(*(const bf8v*)(pb),      *(const bf8v*)(pb + 16));
            v16bf b1 = cat8(*(const bf8v*)(pb + 32), *(const bf8v*)(pb + 48));
            v8f acc = {};
            acc = wmma_bf16(a0, b0, acc);
            acc = wmma_bf16(a1, b1, acc);

            const int c = c0 + am;           // channel 0..127
#pragma unroll
            for (int r = 0; r < 8; ++r) {
                const int m = rb + r;
                const float w  = acc[r] * invH;
                const long  rv = s_recv[m];
                const float es = s_es[m];
                const float e0 = s_ev[m][0], e1 = s_ev[m][1], e2 = s_ev[m][2];
                if (j == 0) {                // m0a -> msg_s[:,0:128]
                    atomicAdd(&msg_s[rv * 256 + c], w * s_xs[m][c] * es);
                } else if (j == 1) {         // m1a -> msg_v rows 0:128
                    float co = w * s_xs[m][c];
                    atomicAdd(&msg_v[             rv * 384 + c], co * e0);
                    atomicAdd(&msg_v[vplane     + rv * 384 + c], co * e1);
                    atomicAdd(&msg_v[2 * vplane + rv * 384 + c], co * e2);
                } else if (j == 2) {         // m1b -> msg_v rows 128:256
                    float co = w * es;
                    atomicAdd(&msg_v[             rv * 384 + 128 + c], co * s_xv[0][m][c]);
                    atomicAdd(&msg_v[vplane     + rv * 384 + 128 + c], co * s_xv[1][m][c]);
                    atomicAdd(&msg_v[2 * vplane + rv * 384 + 128 + c], co * s_xv[2][m][c]);
                } else if (j == 3) {         // m0b -> msg_s[:,128:256]
                    float d = s_xv[0][m][c] * e0 + s_xv[1][m][c] * e1 + s_xv[2][m][c] * e2;
                    atomicAdd(&msg_s[rv * 256 + 128 + c], w * d * is3);
                } else {                     // m1c -> msg_v rows 256:384
                    float x0 = s_xv[0][m][c], x1 = s_xv[1][m][c], x2 = s_xv[2][m][c];
                    atomicAdd(&msg_v[             rv * 384 + 256 + c], w * (x1 * e2 - x2 * e1) * is2);
                    atomicAdd(&msg_v[vplane     + rv * 384 + 256 + c], w * (x2 * e0 - x0 * e2) * is2);
                    atomicAdd(&msg_v[2 * vplane + rv * 384 + 256 + c], w * (x0 * e1 - x1 * e0) * is2);
                }
            }
        }
    }
}

// ---------------------------------------------------------------------------
// Launch
// ---------------------------------------------------------------------------
static inline dim3 flat_grid(long n) { return dim3((unsigned)((n + 255) / 256)); }

extern "C" void kernel_launch(void* const* d_in, const int* in_sizes, int n_in,
                              void* d_out, int out_size, void* d_ws, size_t ws_size,
                              hipStream_t stream) {
    const float* node_attrs   = (const float*)d_in[0];
    const float* node_feats_s = (const float*)d_in[1];
    const float* node_feats_v = (const float*)d_in[2];
    const float* edge_attrs   = (const float*)d_in[3];
    const float* edge_feats   = (const float*)d_in[4];
    const float* W_sc_s       = (const float*)d_in[5];
    const float* W_sc_v       = (const float*)d_in[6];
    const float* W_lin_s      = (const float*)d_in[7];
    const float* W_lin_v      = (const float*)d_in[8];
    const float* mlp_w0       = (const float*)d_in[9];
    const float* mlp_w1       = (const float*)d_in[10];
    const float* mlp_w2       = (const float*)d_in[11];
    const float* mlp_w3       = (const float*)d_in[12];
    const float* W_out_s      = (const float*)d_in[13];
    const float* W_out_v      = (const float*)d_in[14];
    const int*   senders      = (const int*)d_in[15];
    const int*   receivers    = (const int*)d_in[16];

    const int Nn = in_sizes[0] / 10;    // A = 10; N multiple of 16
    const int Ee = in_sizes[15];        // E multiple of 16
    float* out = (float*)d_out;         // [0,N*512): message ; [N*512,2N*512): sc

    // ---- workspace layout ----
    char* wsc = (char*)d_ws;
    size_t off = 0;
    auto alloc = [&](size_t bytes) -> char* {
        char* p = wsc + off; off += (bytes + 63) & ~(size_t)63; return p;
    };
    float*  msg_s  = (float*)alloc((size_t)Nn * (256 + 3 * 384) * 4);
    float*  msg_v  = msg_s + (size_t)Nn * 256;
    __bf16* nfs_bf = (__bf16*)alloc((size_t)Nn * 128 * 2);
    __bf16* nfv_bf = (__bf16*)alloc((size_t)Nn * 384 * 2);     // 3 planes
    __bf16* ef_bf  = (__bf16*)alloc((size_t)Ee * 32 * 2);      // K padded 8->32
    __bf16* xs_bf  = (__bf16*)alloc((size_t)Nn * 128 * 2);
    __bf16* xv_bf  = (__bf16*)alloc((size_t)Nn * 384 * 2);     // 3 planes
    __bf16* hA     = (__bf16*)alloc((size_t)Ee * 64 * 2);
    __bf16* hB     = (__bf16*)alloc((size_t)Ee * 64 * 2);
    __bf16* msg_sb = (__bf16*)alloc((size_t)Nn * (256 + 3 * 384) * 2);
    __bf16* msg_vb = msg_sb + (size_t)Nn * 256;
    __bf16* Wt_lin_s = (__bf16*)alloc(128 * 128 * 2);
    __bf16* Wt_lin_v = (__bf16*)alloc(128 * 128 * 2);
    __bf16* Wt_w0    = (__bf16*)alloc(64 * 32 * 2);
    __bf16* Wt_w1    = (__bf16*)alloc(64 * 64 * 2);
    __bf16* Wt_w2    = (__bf16*)alloc(64 * 64 * 2);
    __bf16* Wt_w3    = (__bf16*)alloc(640 * 64 * 2);
    __bf16* Wt_out_s = (__bf16*)alloc(128 * 256 * 2);
    __bf16* Wt_out_v = (__bf16*)alloc(128 * 384 * 2);
    __bf16* Wt_sc_s  = (__bf16*)alloc(128 * 1280 * 2);
    __bf16* Wt_sc_v  = (__bf16*)alloc(128 * 1280 * 2);

    const int mtN = Nn / 16, mtE = Ee / 16;
    const dim3 blk(32);

    // ---- prep: weight transpose + activation conversion ----
    auto tw = [&](const float* s, __bf16* d, int K, int Nout, int Kpad) {
        transpose_w_kernel<<<flat_grid((long)Nout * Kpad), 256, 0, stream>>>(s, d, K, Nout, Kpad);
    };
    tw(W_lin_s, Wt_lin_s, 128, 128, 128);
    tw(W_lin_v, Wt_lin_v, 128, 128, 128);
    tw(mlp_w0,  Wt_w0,      8,  64,  32);
    tw(mlp_w1,  Wt_w1,     64,  64,  64);
    tw(mlp_w2,  Wt_w2,     64,  64,  64);
    tw(mlp_w3,  Wt_w3,     64, 640,  64);
    tw(W_out_s, Wt_out_s, 256, 128, 256);
    tw(W_out_v, Wt_out_v, 384, 128, 384);
    tw(W_sc_s,  Wt_sc_s, 1280, 128, 1280);
    tw(W_sc_v,  Wt_sc_v, 1280, 128, 1280);

    f32_to_bf16_kernel<<<flat_grid((long)Nn * 128), 256, 0, stream>>>(
        node_feats_s, nfs_bf, (long)Nn * 128);
    planarize_nfv_kernel<<<flat_grid((long)Nn * 384), 256, 0, stream>>>(
        node_feats_v, nfv_bf, Nn);
    pad_rows_bf16_kernel<<<flat_grid((long)Ee * 32), 256, 0, stream>>>(
        edge_feats, ef_bf, Ee, 8, 32);

    hipMemsetAsync(msg_s, 0, (size_t)Nn * (256 + 3 * 384) * sizeof(float), stream);

    const float invc  = 0.08838834765f;   // 1/sqrt(128)
    const float invF  = 0.35355339059f;   // 1/sqrt(8)
    const float invH  = 0.125f;           // 1/sqrt(64)
    const float sOutS = 0.00390625f;      // 1/sqrt(256)/16
    const float sOutV = 0.00318944392f;   // 1/sqrt(384)/16

    // ---- node linears (bf16 out) ----
    gemm_bf16t_kernel<4, true, false><<<dim3(mtN, 8), blk, 0, stream>>>(
        nfs_bf, 128, Wt_lin_s, 128, nullptr, xs_bf, 128, 1, invc);
    for (int i = 0; i < 3; ++i)
        gemm_bf16t_kernel<4, true, false><<<dim3(mtN, 8), blk, 0, stream>>>(
            nfv_bf + (size_t)i * Nn * 128, 128, Wt_lin_v, 128,
            nullptr, xv_bf + (size_t)i * Nn * 128, 128, 1, invc);

    // ---- edge MLP (bf16 chain, fused SiLU) ----
    gemm_bf16t_kernel<1, true, true><<<dim3(mtE, 4), blk, 0, stream>>>(
        ef_bf, 32, Wt_w0, 32, nullptr, hA, 64, 1, invF);
    gemm_bf16t_kernel<2, true, true><<<dim3(mtE, 4), blk, 0, stream>>>(
        hA, 64, Wt_w1, 64, nullptr, hB, 64, 1, invH);
    gemm_bf16t_kernel<2, true, true><<<dim3(mtE, 4), blk, 0, stream>>>(
        hB, 64, Wt_w2, 64, nullptr, hA, 64, 1, invH);

    // ---- self-connection (sc half of output), K=1280, LDS-staged tp ----
    sc_tp_kernel<<<dim3(mtN, 1, 4), dim3(256), 0, stream>>>(
        node_attrs, node_feats_s, node_feats_v, Wt_sc_s, Wt_sc_v,
        out + (size_t)Nn * 512, Nn);

    // ---- fused tpw + message + scatter ----
    edge_msg_kernel<<<dim3(mtE), blk, 0, stream>>>(
        hA, Wt_w3, edge_attrs, senders, receivers, xs_bf, xv_bf,
        msg_s, msg_v, Nn);

    // ---- msg f32 -> bf16 mirror, then output GEMMs (message half) ----
    f32_to_bf16_kernel<<<flat_grid((long)Nn * 1408), 256, 0, stream>>>(
        msg_s, msg_sb, (long)Nn * 1408);

    gemm_bf16t_kernel<8, false, false><<<dim3(mtN, 8), blk, 0, stream>>>(
        msg_sb, 256, Wt_out_s, 256, out, nullptr, 512, 1, sOutS);
    for (int i = 0; i < 3; ++i)
        gemm_bf16t_kernel<12, false, false><<<dim3(mtN, 8), blk, 0, stream>>>(
            msg_vb + (size_t)i * Nn * 384, 384, Wt_out_v, 384,
            out + 128 + i, nullptr, 512, 3, sOutV);
}
